// RealMNISTModel_24730421690961
// MI455X (gfx1250) — matrix operand order repaired and checked by reading, not compiled, not run on hardware
//
#include <hip/hip_runtime.h>
#include <hip/hip_bf16.h>
#include <stdint.h>

typedef __attribute__((ext_vector_type(16))) __bf16 v16bf;
typedef __attribute__((ext_vector_type(8)))  float  v8f;

#define HID    1024
#define DIN    784
#define KPAD   800      // 25 k-steps of 32
#define KSTEPS 25
#define NCLS   10
#define NP     16       // padded output cols (one WMMA N tile)
#define MT     64       // rows per workgroup
#define BROWS  16384

// ---------------- prep 1: M1t[16][800] bf16 = (W_in @ diag(g) W_out)^T, w_mean[800] ----------------
__global__ __launch_bounds__(256) void prep_m1(const float* __restrict__ Win,
                                               const float* __restrict__ Wout,
                                               __bf16* __restrict__ M1t,
                                               float* __restrict__ wmean) {
  const int k = blockIdx.x;              // 0..799
  __shared__ float part[16][17];
  __shared__ float wpart[16];
  const int t = threadIdx.x, c = t & 15, js = t >> 4;
  float s = 0.f;
  if (k < DIN) {
    for (int j = js * 64; j < js * 64 + 64; ++j) {
      float g = (j < 5) ? 2.f : 1.f;
      float w = (c < NCLS) ? Wout[j * NCLS + c] : 0.f;
      s = fmaf(Win[k * HID + j], g * w, s);
    }
  }
  part[js][c] = s;
  if (c == 0) {
    float sm = 0.f;
    if (k < DIN)
      for (int j = js * 64; j < js * 64 + 64; ++j) sm += Win[k * HID + j];
    wpart[js] = sm;
  }
  __syncthreads();
  if (t < 16) {
    float acc = 0.f;
    for (int q = 0; q < 16; ++q) acc += part[q][t];
    M1t[t * KPAD + k] = (__bf16)acc;
  }
  if (t == 16) {
    float acc = 0.f;
    for (int q = 0; q < 16; ++q) acc += wpart[q];
    wmean[k] = acc * (1.0f / 1024.0f);
  }
}

// ---------------- prep 2: M2[64][16], cb[16], c_mean ----------------
__global__ __launch_bounds__(1024) void prep_m2(const float* __restrict__ Wph,
                                                const float* __restrict__ Wout,
                                                const float* __restrict__ b_in,
                                                const float* __restrict__ b_ph,
                                                const float* __restrict__ b_out,
                                                float* __restrict__ M2,
                                                float* __restrict__ cb,
                                                float* __restrict__ cmean) {
  const int t = threadIdx.x, h = t >> 4, c = t & 15;   // h:0..63, c:0..15
  float s = 0.f;
  for (int j = 0; j < HID; ++j) {
    float g = (j < 5) ? 2.f : 1.f;
    float w = (c < NCLS) ? Wout[j * NCLS + c] : 0.f;
    s = fmaf(Wph[h * HID + j], g * w, s);
  }
  M2[h * NP + c] = s;
  if (h == 0) {
    float sb = (c < NCLS) ? b_out[c] : 0.f;
    for (int j = 0; j < HID; ++j) {
      float g = (j < 5) ? 2.f : 1.f;
      float w = (c < NCLS) ? Wout[j * NCLS + c] : 0.f;
      sb = fmaf(b_in[j] + b_ph[j], g * w, sb);
    }
    cb[c] = sb;
    if (c == 0) {
      float sm = 0.f;
      for (int j = 0; j < HID; ++j) sm += b_in[j];
      *cmean = sm * (1.0f / 1024.0f);
    }
  }
}

// ---------------- fused main: out = x@M1 + phasor(x@w_mean + cm)@M2 + cb ----------------
__global__ __launch_bounds__(128) void fused_main(const float* __restrict__ x,
                                                  const __bf16* __restrict__ M1t,
                                                  const float* __restrict__ wmean,
                                                  const float* __restrict__ M2,
                                                  const float* __restrict__ cb,
                                                  const float* __restrict__ cmean,
                                                  float* __restrict__ out) {
  __shared__ __align__(16) float As[2][MT][32];       // double-buffered fp32 A tiles
  __shared__ __align__(32) __bf16 Bs[16][KPAD];       // M1 transposed, bf16
  __shared__ __align__(16) float wms[KPAD];
  __shared__ __align__(16) float M2s[64][NP];
  __shared__ float cbs[NP];
  __shared__ float cms;
  __shared__ float mpart[MT][2];
  __shared__ float out2[MT][NP];

  const int t = threadIdx.x;
  const int lane = t & 31, wave = t >> 5;
  const int n = lane & 15, khalf = lane >> 4;
  const int row0 = blockIdx.x * MT;

  // stage folded constants once (16B vector copies)
  {
    const uint4* s1 = (const uint4*)M1t;     uint4* d1 = (uint4*)&Bs[0][0];
    for (int i = t; i < (16 * KPAD * 2) / 16; i += 128) d1[i] = s1[i];
    const uint4* s2 = (const uint4*)wmean;   uint4* d2 = (uint4*)&wms[0];
    for (int i = t; i < (KPAD * 4) / 16; i += 128) d2[i] = s2[i];
    const uint4* s3 = (const uint4*)M2;      uint4* d3 = (uint4*)&M2s[0][0];
    for (int i = t; i < (64 * NP * 4) / 16; i += 128) d3[i] = s3[i];
    if (t < NP) cbs[t] = cb[t];
    if (t == 0) cms = *cmean;
  }

  // async staging helpers: 4 instructions per wave per tile (ASYNCcnt += 4)
  auto stage_full = [&](int kt, int b) {
    #pragma unroll
    for (int i = 0; i < 4; ++i) {
      const int seg = t + 128 * i, r = seg >> 3, ss = seg & 7;
      uint32_t lo = (uint32_t)(uintptr_t)&As[b][r][4 * ss];
      const float* ga = x + (size_t)(row0 + r) * DIN + (size_t)kt * 32 + 4 * ss;
      asm volatile("global_load_async_to_lds_b128 %0, %1, off"
                   :: "v"(lo), "v"(ga) : "memory");
    }
  };
  auto stage_last = [&](int b) {                      // tile 24: K 768..783 valid, 784..799 zero
    #pragma unroll
    for (int i = 0; i < 4; ++i) {
      const int seg = t + 128 * i, r = seg >> 3, ss = seg & 7;
      if (ss < 4) {
        uint32_t lo = (uint32_t)(uintptr_t)&As[b][r][4 * ss];
        const float* ga = x + (size_t)(row0 + r) * DIN + 768 + 4 * ss;
        asm volatile("global_load_async_to_lds_b128 %0, %1, off"
                     :: "v"(lo), "v"(ga) : "memory");
      } else {
        float4 z = {0.f, 0.f, 0.f, 0.f};
        *(float4*)&As[b][r][4 * ss] = z;
      }
    }
  };

  stage_full(0, 0);                                   // prologue prefetch

  v8f acc = {};
  float macc = 0.f;
  const int mrow = t >> 1, mhalf = t & 1;             // mean path: 2 threads/row
  const int am = 16 * wave + n;                       // A fragment row for this lane

  for (int it = 0; it < KSTEPS; ++it) {
    const int b = it & 1;
    const int k0 = it * 32;

    __syncthreads();                                  // compute(it-1) done: safe to overwrite buf[!b]
    if (it + 1 < KSTEPS) {
      if (it + 1 == KSTEPS - 1) stage_last(1 - b);
      else                      stage_full(it + 1, 1 - b);
      asm volatile("s_wait_asynccnt 4" ::: "memory"); // <=4 outstanding => tile it complete
    } else {
      asm volatile("s_wait_asynccnt 0" ::: "memory");
    }
    __syncthreads();                                  // tile it visible to all waves

    // A fragment: 16 contiguous K per lane half (k = 16*khalf + s), matching B
    const float* ap = &As[b][am][0];
    float4 q0 = *(const float4*)(ap + 16 * khalf);
    float4 q1 = *(const float4*)(ap + 16 * khalf + 4);
    float4 q2 = *(const float4*)(ap + 16 * khalf + 8);
    float4 q3 = *(const float4*)(ap + 16 * khalf + 12);
    v16bf a;
    a[0]=(__bf16)q0.x; a[1]=(__bf16)q0.y; a[2]=(__bf16)q0.z; a[3]=(__bf16)q0.w;
    a[4]=(__bf16)q1.x; a[5]=(__bf16)q1.y; a[6]=(__bf16)q1.z; a[7]=(__bf16)q1.w;
    a[8]=(__bf16)q2.x; a[9]=(__bf16)q2.y; a[10]=(__bf16)q2.z; a[11]=(__bf16)q2.w;
    a[12]=(__bf16)q3.x; a[13]=(__bf16)q3.y; a[14]=(__bf16)q3.z; a[15]=(__bf16)q3.w;

    // B fragment: 16 contiguous bf16 of column n (this code's k order: lane half h holds K 16h..16h+15)
    v16bf bb = *(const v16bf*)&Bs[n][k0 + 16 * khalf];

    acc = __builtin_amdgcn_wmma_f32_16x16x32_bf16(false, a, false, bb,
                                                  (short)0, acc, false, false);

    // exact fp32 mean path (phase is amplified x224 downstream)
    #pragma unroll
    for (int kk = 0; kk < 16; ++kk)
      macc = fmaf(As[b][mrow][16 * mhalf + kk], wms[k0 + 16 * mhalf + kk], macc);
  }

  mpart[mrow][mhalf] = macc;
  __syncthreads();

  // phasor + temporal@M2 per row (one thread per row)
  if (t < MT) {
    const float xm = mpart[t][0] + mpart[t][1] + cms;
    float o[NP];
    #pragma unroll
    for (int c = 0; c < NP; ++c) o[c] = cbs[c];
    for (int h = 0; h < 32; ++h) {
      float ph = (float)(h + 1) * 7.0f * xm;
      float sn, cs;
      __sincosf(ph, &sn, &cs);
      #pragma unroll
      for (int c = 0; c < NP; ++c)
        o[c] = fmaf(cs, M2s[h][c], fmaf(sn, M2s[32 + h][c], o[c]));
    }
    #pragma unroll
    for (int c = 0; c < NP; ++c) out2[t][c] = o[c];
  }
  __syncthreads();

  // epilogue: C layout — VGPR r holds M = r + 8*khalf, N = lane%16
  #pragma unroll
  for (int r = 0; r < 8; ++r) {
    const int mloc = 16 * wave + r + 8 * khalf;
    const float v = acc[r] + out2[mloc][n];
    if (n < NCLS) out[(size_t)(row0 + mloc) * NCLS + n] = v;
  }
}

extern "C" void kernel_launch(void* const* d_in, const int* in_sizes, int n_in,
                              void* d_out, int out_size, void* d_ws, size_t ws_size,
                              hipStream_t stream) {
  (void)in_sizes; (void)n_in; (void)out_size; (void)ws_size;
  const float* x     = (const float*)d_in[0];
  const float* W_in  = (const float*)d_in[1];
  const float* b_in  = (const float*)d_in[2];
  const float* W_ph  = (const float*)d_in[3];
  const float* b_ph  = (const float*)d_in[4];
  const float* W_out = (const float*)d_in[5];
  const float* b_out = (const float*)d_in[6];
  float* out = (float*)d_out;

  char* ws = (char*)d_ws;
  __bf16* M1t  = (__bf16*)ws;              // 16*800*2   = 25600 B
  float*  wm   = (float*)(ws + 25600);     // 800*4      = 3200 B
  float*  M2   = (float*)(ws + 28800);     // 64*16*4    = 4096 B
  float*  cb   = (float*)(ws + 32896);     // 16*4       = 64 B
  float*  cm   = (float*)(ws + 32960);     // 4 B

  prep_m1<<<dim3(KPAD), dim3(256), 0, stream>>>(W_in, W_out, M1t, wm);
  prep_m2<<<dim3(1), dim3(1024), 0, stream>>>(W_ph, W_out, b_in, b_ph, b_out, M2, cb, cm);
  fused_main<<<dim3(BROWS / MT), dim3(128), 0, stream>>>(x, M1t, wm, M2, cb, cm, out);
}
